// SynthesisBlock_11355893531017
// MI455X (gfx1250) — matrix operand order repaired and checked by reading, not compiled, not run on hardware
//
#include <hip/hip_runtime.h>

// ---------------------------------------------------------------------------
// CDNA5 (gfx1250) implementation of the StyleGAN2-like mesh SynthesisBlock.
// Heavy work = two modulated face-convs mapped onto v_wmma_f32_16x16x32_f16,
// with 9-neighborhood tiles staged through LDS once per workgroup.
// ---------------------------------------------------------------------------

typedef _Float16 v16h __attribute__((ext_vector_type(16)));
typedef _Float16 v8h  __attribute__((ext_vector_type(8)));
typedef float    v8f  __attribute__((ext_vector_type(8)));

union H16 { v16h v; v8h h[2]; };

constexpr int kFc   = 32768;     // coarse faces
constexpr int kFf   = 131072;    // fine faces
constexpr int kPad0 = 1024;
constexpr int kPad1 = 4096;
constexpr int kIn   = 128;       // conv0 input channels
constexpr int kOut  = 128;       // conv0 output channels / conv1 input
constexpr int kOc1  = 96;        // conv1 output channels
constexpr int kWd   = 512;       // w latent dim

constexpr float kGain  = 1.41421356237309515f;  // sqrt(2) lrelu def_gain
constexpr float kClamp = 256.0f;
constexpr float kSlope = 0.2f;

__device__ __forceinline__ float act_clamp(float v) {
  v = (v < 0.f ? kSlope * v : v) * kGain;
  return fminf(fmaxf(v, -kClamp), kClamp);
}

// ---------------------------------------------------------------------------
// Kernel 1: style affines + weight modulation/demodulation (tiny, 1 block).
//   s = w @ A.T / sqrt(512) + b ;  Wmod = W * s (f16) ; d = rsqrt(sum Wmod^2)
// ---------------------------------------------------------------------------
__global__ void __launch_bounds__(128)
prep_kernel(const float* __restrict__ wsv,
            const float* __restrict__ a0w, const float* __restrict__ a0b,
            const float* __restrict__ wc0, const float* __restrict__ wsd0,
            const float* __restrict__ wcr0,
            const float* __restrict__ a1w, const float* __restrict__ a1b,
            const float* __restrict__ wc1, const float* __restrict__ wsd1,
            const float* __restrict__ wcr1,
            const float* __restrict__ a2w, const float* __restrict__ a2b,
            const float* __restrict__ wrgb,
            float* __restrict__ d0v, float* __restrict__ d1v,
            float* __restrict__ wrgbm,
            _Float16* __restrict__ Wc0h, _Float16* __restrict__ Wsd0h,
            _Float16* __restrict__ Wcr0h,
            _Float16* __restrict__ Wc1h, _Float16* __restrict__ Wsd1h,
            _Float16* __restrict__ Wcr1h)
{
  __shared__ float s0[kIn], s1[kOut], s2[kOc1];
  const int j = threadIdx.x;                       // 0..127
  const float invSqWd = 0.044194173824159216f;     // 1/sqrt(512)
  const float invSqOc = 0.10206207261596577f;      // 1/sqrt(96)

  float a0 = 0.f, a1 = 0.f, a2 = 0.f;
  for (int w = 0; w < kWd; ++w) {
    a0 += wsv[w]          * a0w[j * kWd + w];
    a1 += wsv[kWd + w]    * a1w[j * kWd + w];
    if (j < kOc1) a2 += wsv[2 * kWd + w] * a2w[j * kWd + w];
  }
  s0[j] = a0 * invSqWd + a0b[j];
  s1[j] = a1 * invSqWd + a1b[j];
  if (j < kOc1) s2[j] = (a2 * invSqWd + a2b[j]) * invSqOc;
  __syncthreads();

  { // conv0 modulated weights, output row o = j
    float sc = 0.f, ss = 0.f, sr = 0.f;
    for (int i = 0; i < kIn; ++i) {
      float si = s0[i];
      float mc = wc0[j * kIn + i] * si;
      float ms = wsd0[j * kIn + i] * si;
      float mr = wcr0[j * kIn + i] * si;
      sc += mc * mc; ss += ms * ms; sr += mr * mr;
      Wc0h[j * kIn + i]  = (_Float16)mc;
      Wsd0h[j * kIn + i] = (_Float16)ms;
      Wcr0h[j * kIn + i] = (_Float16)mr;
    }
    d0v[j] = rsqrtf(sc + 4.f * ss + 4.f * sr + 1e-8f);  // 1 center + 4 side + 4 corner
  }
  if (j < kOc1) {
    float sc = 0.f, ss = 0.f, sr = 0.f;
    for (int i = 0; i < kOut; ++i) {
      float si = s1[i];
      float mc = wc1[j * kOut + i] * si;
      float ms = wsd1[j * kOut + i] * si;
      float mr = wcr1[j * kOut + i] * si;
      sc += mc * mc; ss += ms * ms; sr += mr * mr;
      Wc1h[j * kOut + i]  = (_Float16)mc;
      Wsd1h[j * kOut + i] = (_Float16)ms;
      Wcr1h[j * kOut + i] = (_Float16)mr;
    }
    d1v[j] = rsqrtf(sc + 4.f * ss + 4.f * sr + 1e-8f);
  }
  if (j < 3) {
    for (int i = 0; i < kOc1; ++i) wrgbm[j * kOc1 + i] = wrgb[j * kOc1 + i] * s2[i];
  }
}

// ---------------------------------------------------------------------------
// Kernel 2: concat(x, shape_feat) -> f16 padded table [(Fc+pad0) x 128]
// ---------------------------------------------------------------------------
__global__ void __launch_bounds__(256)
concat_pad_kernel(const float* __restrict__ x, const float* __restrict__ sfeat,
                  _Float16* __restrict__ h0pad)
{
  const int t    = threadIdx.x;
  const int face = blockIdx.x * 8 + (t >> 5);
  const int c0   = (t & 31) * 4;
  _Float16* dst  = h0pad + (size_t)face * kIn + c0;
  if (face >= kFc) {
    dst[0] = dst[1] = dst[2] = dst[3] = (_Float16)0.f;
    return;
  }
  float v0, v1, v2, v3;
  if (c0 < 96) {
    const float* s = x + (size_t)face * 96 + c0;
    v0 = s[0]; v1 = s[1]; v2 = s[2]; v3 = s[3];
  } else {
    const float* s = sfeat + (size_t)face * 32 + (c0 - 96);
    v0 = s[0]; v1 = s[1]; v2 = s[2]; v3 = s[3];
  }
  dst[0] = (_Float16)v0; dst[1] = (_Float16)v1;
  dst[2] = (_Float16)v2; dst[3] = (_Float16)v3;
}

// ---------------------------------------------------------------------------
// Shared conv machinery: stage 9x16x128 f16 neighborhood tile in LDS, then
// per-wave WMMA over K=128 for each of 9 positions with running f32 max.
// ---------------------------------------------------------------------------
__device__ __forceinline__ void
load_tile_lds(_Float16* __restrict__ smA, const _Float16* __restrict__ hpad,
              const long long* __restrict__ fn, long long faceBase)
{
  const int t  = threadIdx.x;
  const int fr = t >> 4;          // face row in tile 0..15
  const int cc = (t & 15) * 8;    // 8-half chunk within 128-ch row
#pragma unroll
  for (int pos = 0; pos < 9; ++pos) {
    long long g = fn[(faceBase + fr) * 9 + pos];   // gathered (padded) row index
    *(v8h*)(smA + (size_t)(pos * 16 + fr) * kIn + cc) =
        *(const v8h*)(hpad + (size_t)g * kIn + cc);
  }
}

__device__ __forceinline__ v8f
conv_wave_compute(const _Float16* __restrict__ smA, int lane,
                  const _Float16* __restrict__ Wc,
                  const _Float16* __restrict__ Wsd,
                  const _Float16* __restrict__ Wcr, int orow)
{
  const int fr = lane & 15;       // A row M = lane%16
  const int hi = lane >> 4;       // lane half selects K sub-chunks
  v8f runmax;
#pragma unroll
  for (int j = 0; j < 8; ++j) runmax[j] = -3.4e38f;

  const _Float16* Wt[3]  = {Wc, Wsd, Wcr};
  const int pstart[3]    = {0, 1, 5};
  const int pcnt[3]      = {1, 4, 4};

  for (int ty = 0; ty < 3; ++ty) {
    // B fragments for the 4 K-steps: lane holds N=orow, 16 contiguous K at hi*16
    const _Float16* wbase = Wt[ty] + (size_t)orow * kIn + hi * 16;
    H16 b[4];
#pragma unroll
    for (int kt = 0; kt < 4; ++kt) b[kt].v = *(const v16h*)(wbase + kt * 32);

    for (int p = 0; p < pcnt[ty]; ++p) {
      const int pos = pstart[ty] + p;
      const _Float16* arow = smA + (size_t)(pos * 16 + fr) * kIn;
      v8f acc;
#pragma unroll
      for (int j = 0; j < 8; ++j) acc[j] = 0.f;
#pragma unroll
      for (int kt = 0; kt < 4; ++kt) {
        H16 a;   // A 16x32: 16 contiguous K at kt*32 + hi*16
        a.h[0] = *(const v8h*)(arow + kt * 32 + hi * 16);
        a.h[1] = *(const v8h*)(arow + kt * 32 + hi * 16 + 8);
        acc = __builtin_amdgcn_wmma_f32_16x16x32_f16(
            false, a.v, false, b[kt].v, (short)0, acc, false, false);
      }
#pragma unroll
      for (int j = 0; j < 8; ++j) runmax[j] = fmaxf(runmax[j], acc[j]);
    }
  }
  return runmax;
}

// ---------------------------------------------------------------------------
// Kernel 3: conv0 (coarse, 128->128), epilogue = demod scale, f32 out.
// One block = 16 faces; 8 waves, wave w owns output tile N = [16w, 16w+16).
// ---------------------------------------------------------------------------
__global__ void __launch_bounds__(256)
conv0_kernel(const _Float16* __restrict__ hpad, const long long* __restrict__ fn,
             const _Float16* __restrict__ Wc, const _Float16* __restrict__ Wsd,
             const _Float16* __restrict__ Wcr, const float* __restrict__ dvec,
             float* __restrict__ out)
{
  __shared__ __align__(16) _Float16 smA[9 * 16 * kIn];
  const long long faceBase = (long long)blockIdx.x * 16;
  load_tile_lds(smA, hpad, fn, faceBase);
  __syncthreads();

  const int wave = threadIdx.x >> 5, lane = threadIdx.x & 31;
  const int hi   = lane >> 4;
  const int orow = wave * 16 + (lane & 15);
  v8f rm = conv_wave_compute(smA, lane, Wc, Wsd, Wcr, orow);
  const float dsc = dvec[orow];
#pragma unroll
  for (int j = 0; j < 8; ++j) {
    long long f = faceBase + hi * 8 + j;    // D: M = j + 8*hi
    out[f * kOut + orow] = rm[j] * dsc;
  }
}

// ---------------------------------------------------------------------------
// Kernel 4: smooth-upsample (mean over 9 coarse neighbors via pool_map) +
// noise + bias + lrelu*sqrt(2) + clamp -> padded f16 fine table.
// ---------------------------------------------------------------------------
__global__ void __launch_bounds__(256)
upsample_act_kernel(const float* __restrict__ hc0,
                    const long long* __restrict__ pool_map,
                    const long long* __restrict__ fn1,
                    const float* __restrict__ noise_const,
                    const float* __restrict__ noise_strength,
                    const float* __restrict__ bias,
                    _Float16* __restrict__ h1pad)
{
  const int t    = threadIdx.x;
  const int face = blockIdx.x * 8 + (t >> 5);
  const int c4   = (t & 31);               // float4 index within 128-ch row
  _Float16* dst  = h1pad + (size_t)face * kOut + c4 * 4;
  if (face >= kFf) {                        // zero padding rows
    dst[0] = dst[1] = dst[2] = dst[3] = (_Float16)0.f;
    return;
  }
  float a0 = 0.f, a1 = 0.f, a2 = 0.f, a3 = 0.f;
  for (int k = 0; k < 9; ++k) {
    long long j = fn1[(size_t)face * 9 + k];
    if (j < kFf) {
      const float4* r = (const float4*)(hc0 + (size_t)pool_map[j] * kOut) + c4;
      float4 rv = *r;
      a0 += rv.x; a1 += rv.y; a2 += rv.z; a3 += rv.w;
    }
  }
  const float inv9 = 1.f / 9.f;
  const float nz   = noise_const[face] * noise_strength[0];
  const int   c0   = c4 * 4;
  dst[0] = (_Float16)act_clamp(a0 * inv9 + nz + bias[c0 + 0]);
  dst[1] = (_Float16)act_clamp(a1 * inv9 + nz + bias[c0 + 1]);
  dst[2] = (_Float16)act_clamp(a2 * inv9 + nz + bias[c0 + 2]);
  dst[3] = (_Float16)act_clamp(a3 * inv9 + nz + bias[c0 + 3]);
}

// ---------------------------------------------------------------------------
// Kernel 5: conv1 (fine, 128->96) with fused noise + act epilogue -> d_out h.
// 6 active waves (96 outputs); waves 6,7 only help with the LDS staging.
// ---------------------------------------------------------------------------
__global__ void __launch_bounds__(256)
conv1_kernel(const _Float16* __restrict__ hpad, const long long* __restrict__ fn,
             const _Float16* __restrict__ Wc, const _Float16* __restrict__ Wsd,
             const _Float16* __restrict__ Wcr, const float* __restrict__ dvec,
             const float* __restrict__ noise_const,
             const float* __restrict__ noise_strength,
             const float* __restrict__ bias, float* __restrict__ out)
{
  __shared__ __align__(16) _Float16 smA[9 * 16 * kOut];
  const long long faceBase = (long long)blockIdx.x * 16;
  load_tile_lds(smA, hpad, fn, faceBase);
  __syncthreads();

  const int wave = threadIdx.x >> 5, lane = threadIdx.x & 31;
  if (wave < 6) {                               // 6 * 16 = 96 output channels
    const int hi   = lane >> 4;
    const int orow = wave * 16 + (lane & 15);
    v8f rm = conv_wave_compute(smA, lane, Wc, Wsd, Wcr, orow);
    const float dsc = dvec[orow];
    const float ns  = noise_strength[0];
    const float bo  = bias[orow];
#pragma unroll
    for (int j = 0; j < 8; ++j) {
      long long f = faceBase + hi * 8 + j;
      float v = rm[j] * dsc + noise_const[f] * ns + bo;
      out[f * kOc1 + orow] = act_clamp(v);
    }
  }
}

// ---------------------------------------------------------------------------
// Kernel 6: torgb (1x1 modulated conv, demod off, gather fn1[:,0]) +
// img smooth-upsample -> d_out img tail.
// ---------------------------------------------------------------------------
__global__ void __launch_bounds__(256)
torgb_kernel(const float* __restrict__ hout, const long long* __restrict__ fn1,
             const long long* __restrict__ pool_map, const float* __restrict__ img,
             const float* __restrict__ wrgbm, const float* __restrict__ brgb,
             float* __restrict__ img_out)
{
  const int f = blockIdx.x * blockDim.x + threadIdx.x;
  if (f >= kFf) return;

  float y0 = 0.f, y1 = 0.f, y2 = 0.f;
  long long j0 = fn1[(size_t)f * 9];
  if (j0 < kFf) {
    const float* hr = hout + (size_t)j0 * kOc1;
    for (int i = 0; i < kOc1; ++i) {
      float hv = hr[i];
      y0 += hv * wrgbm[i];
      y1 += hv * wrgbm[kOc1 + i];
      y2 += hv * wrgbm[2 * kOc1 + i];
    }
  }
  y0 = fminf(fmaxf(y0 + brgb[0], -kClamp), kClamp);
  y1 = fminf(fmaxf(y1 + brgb[1], -kClamp), kClamp);
  y2 = fminf(fmaxf(y2 + brgb[2], -kClamp), kClamp);

  float m0 = 0.f, m1 = 0.f, m2 = 0.f;
  for (int k = 0; k < 9; ++k) {
    long long j = fn1[(size_t)f * 9 + k];
    if (j < kFf) {
      const float* r = img + (size_t)pool_map[j] * 3;
      m0 += r[0]; m1 += r[1]; m2 += r[2];
    }
  }
  const float inv9 = 1.f / 9.f;
  img_out[(size_t)f * 3 + 0] = m0 * inv9 + y0;
  img_out[(size_t)f * 3 + 1] = m1 * inv9 + y1;
  img_out[(size_t)f * 3 + 2] = m2 * inv9 + y2;
}

// ---------------------------------------------------------------------------
extern "C" void kernel_launch(void* const* d_in, const int* in_sizes, int n_in,
                              void* d_out, int out_size, void* d_ws, size_t ws_size,
                              hipStream_t stream)
{
  (void)in_sizes; (void)n_in; (void)out_size; (void)ws_size;

  const float*     x     = (const float*)d_in[0];
  const float*     sfeat = (const float*)d_in[1];
  const float*     img   = (const float*)d_in[2];
  const float*     wsv   = (const float*)d_in[3];
  const long long* fn0   = (const long long*)d_in[4];
  const long long* fn1   = (const long long*)d_in[5];
  const long long* pm0   = (const long long*)d_in[6];
  // d_in[7], d_in[8] are pad0/pad1 scalars (compile-time constants here)
  const float* a0w  = (const float*)d_in[9];
  const float* a0b  = (const float*)d_in[10];
  const float* wc0  = (const float*)d_in[11];
  const float* wsd0 = (const float*)d_in[12];
  const float* wcr0 = (const float*)d_in[13];
  const float* ns0  = (const float*)d_in[14];
  const float* b0   = (const float*)d_in[15];
  const float* nc0  = (const float*)d_in[16];
  const float* a1w  = (const float*)d_in[17];
  const float* a1b  = (const float*)d_in[18];
  const float* wc1  = (const float*)d_in[19];
  const float* wsd1 = (const float*)d_in[20];
  const float* wcr1 = (const float*)d_in[21];
  const float* ns1  = (const float*)d_in[22];
  const float* b1   = (const float*)d_in[23];
  const float* nc1  = (const float*)d_in[24];
  const float* a2w  = (const float*)d_in[25];
  const float* a2b  = (const float*)d_in[26];
  const float* wrgb = (const float*)d_in[27];
  const float* brgb = (const float*)d_in[28];

  // --- scratch carve-out (all offsets 256B-aligned, ~60 MB total) ---
  char* p = (char*)d_ws;
  auto take = [&](size_t bytes) {
    char* r = p;
    p += (bytes + 255) & ~size_t(255);
    return r;
  };
  float*    d0v   = (float*)take(kOut * sizeof(float));
  float*    d1v   = (float*)take(kOc1 * sizeof(float));
  float*    wrgbm = (float*)take(3 * kOc1 * sizeof(float));
  _Float16* Wc0h  = (_Float16*)take((size_t)kOut * kIn * 2);
  _Float16* Wsd0h = (_Float16*)take((size_t)kOut * kIn * 2);
  _Float16* Wcr0h = (_Float16*)take((size_t)kOut * kIn * 2);
  _Float16* Wc1h  = (_Float16*)take((size_t)kOc1 * kOut * 2);
  _Float16* Wsd1h = (_Float16*)take((size_t)kOc1 * kOut * 2);
  _Float16* Wcr1h = (_Float16*)take((size_t)kOc1 * kOut * 2);
  _Float16* h0pad = (_Float16*)take((size_t)(kFc + kPad0) * kIn * 2);
  float*    hc0   = (float*)take((size_t)kFc * kOut * 4);
  _Float16* h1pad = (_Float16*)take((size_t)(kFf + kPad1) * kOut * 2);

  float* out_h   = (float*)d_out;                  // [Ff, 96]
  float* out_img = out_h + (size_t)kFf * kOc1;     // [Ff, 3]

  prep_kernel<<<1, 128, 0, stream>>>(
      wsv, a0w, a0b, wc0, wsd0, wcr0, a1w, a1b, wc1, wsd1, wcr1,
      a2w, a2b, wrgb, d0v, d1v, wrgbm,
      Wc0h, Wsd0h, Wcr0h, Wc1h, Wsd1h, Wcr1h);

  concat_pad_kernel<<<(kFc + kPad0) / 8, 256, 0, stream>>>(x, sfeat, h0pad);

  conv0_kernel<<<kFc / 16, 256, 0, stream>>>(
      h0pad, fn0, Wc0h, Wsd0h, Wcr0h, d0v, hc0);

  upsample_act_kernel<<<(kFf + kPad1) / 8, 256, 0, stream>>>(
      hc0, pm0, fn1, nc0, ns0, b0, h1pad);

  conv1_kernel<<<kFf / 16, 256, 0, stream>>>(
      h1pad, fn1, Wc1h, Wsd1h, Wcr1h, d1v, nc1, ns1, b1, out_h);

  torgb_kernel<<<kFf / 256, 256, 0, stream>>>(
      out_h, fn1, pm0, img, wrgbm, brgb, out_img);
}
